// SimpleDecoder_20469814133171
// MI455X (gfx1250) — compile-verified, matrix operand order checked by reading, not run on hardware
//
#include <hip/hip_runtime.h>
#include <hip/hip_bf16.h>
#include <math.h>

#define BS 256
#define FEAT 2048
#define EMB 512
#define MEM 1024
#define VOCAB 10000
#define TLEN 32
#define NSTEP 31

#define LSTRIDE 72   // LDS row stride (halfwords) for 64-wide K slabs, padded for banks
#define NTILE_OUT 5  // N tiles per block in logits GEMM (625 = 5 * 125)

typedef unsigned short u16;
typedef __attribute__((ext_vector_type(16))) __bf16 v16bf;
typedef __attribute__((ext_vector_type(8)))  float  v8f;

union Frag {
  uint4 u[2];
  v16bf v;
};

__device__ __forceinline__ u16 f2bf(float f) {
  unsigned int u = __float_as_uint(f);
  u += 0x7FFFu + ((u >> 16) & 1u);   // round-to-nearest-even
  return (u16)(u >> 16);
}

__device__ __forceinline__ float sigmoidf_(float x) {
  return 1.0f / (1.0f + __expf(-x));
}

__device__ __forceinline__ v8f wmma_bf16(const Frag& a, const Frag& b, v8f c) {
  return __builtin_amdgcn_wmma_f32_16x16x32_bf16(false, a.v, false, b.v,
                                                 (short)0, c, false, false);
}

// ---------------------------------------------------------------- converts
__global__ void f32_to_bf16_k(const float* __restrict__ src, u16* __restrict__ dst, int n) {
  int i = blockIdx.x * blockDim.x + threadIdx.x;
  int stride = gridDim.x * blockDim.x;
  for (; i < n; i += stride) dst[i] = f2bf(src[i]);
}

// gather embeddings for all 31 steps into bf16: x[t][b][e] = emb[caption[b][t]][e]
__global__ void gather_x_k(const float* __restrict__ emb, const int* __restrict__ cap,
                           u16* __restrict__ xbf, int total) {
  int i = blockIdx.x * blockDim.x + threadIdx.x;
  int stride = gridDim.x * blockDim.x;
  for (; i < total; i += stride) {
    int e = i & (EMB - 1);
    int b = (i >> 9) & (BS - 1);
    int t = i >> 17;                       // EMB*BS = 2^17
    int tok = cap[b * TLEN + t];
    xbf[i] = f2bf(emb[(size_t)tok * EMB + e]);
  }
}

__global__ void zero_k(float* p, int n) {
  int i = blockIdx.x * blockDim.x + threadIdx.x;
  if (i < n) p[i] = 0.0f;
}

// ---------------------------------------------------------------- init GEMMs (run once)
// Register-blocked C = A * W^T + bias, all operands from global/L2.
template <int MB, int NB>
__global__ void gemm_bf16_k(const u16* __restrict__ A, int lda,
                            const u16* __restrict__ W, int ldw,
                            const float* __restrict__ bias,
                            float* __restrict__ Cf, u16* __restrict__ Cbf, int ldc,
                            int Mgroups, int Ngroups, int K) {
  int wid = blockIdx.x * (blockDim.x >> 5) + (threadIdx.x >> 5);
  if (wid >= Mgroups * Ngroups) return;
  int mg = wid % Mgroups;
  int ng = wid / Mgroups;
  int lane = threadIdx.x & 31;
  int half = lane >> 4;
  int l = lane & 15;

  const u16* Ap[MB];
  const u16* Wp[NB];
#pragma unroll
  for (int i = 0; i < MB; ++i)
    Ap[i] = A + (size_t)((mg * MB + i) * 16 + l) * lda + half * 8;
#pragma unroll
  for (int j = 0; j < NB; ++j)
    Wp[j] = W + (size_t)((ng * NB + j) * 16 + l) * ldw + half * 8;

  v8f zero = {};
  v8f acc[MB][NB];
#pragma unroll
  for (int i = 0; i < MB; ++i)
#pragma unroll
    for (int j = 0; j < NB; ++j) acc[i][j] = zero;

  for (int k0 = 0; k0 < K; k0 += 32) {
    Frag a[MB], b[NB];
#pragma unroll
    for (int i = 0; i < MB; ++i) {
      const uint4* pa = (const uint4*)(Ap[i] + k0);
      a[i].u[0] = pa[0]; a[i].u[1] = pa[2];
    }
#pragma unroll
    for (int j = 0; j < NB; ++j) {
      const uint4* pb = (const uint4*)(Wp[j] + k0);
      b[j].u[0] = pb[0]; b[j].u[1] = pb[2];
    }
#pragma unroll
    for (int i = 0; i < MB; ++i)
#pragma unroll
      for (int j = 0; j < NB; ++j)
        acc[i][j] = wmma_bf16(a[i], b[j], acc[i][j]);
  }

#pragma unroll
  for (int j = 0; j < NB; ++j) {
    int ncol = (ng * NB + j) * 16 + l;
    float bb = bias[ncol];
#pragma unroll
    for (int i = 0; i < MB; ++i) {
#pragma unroll
      for (int r = 0; r < 8; ++r) {
        int mr = (mg * MB + i) * 16 + r + 8 * half;
        float v = acc[i][j][r] + bb;
        if (Cf)  Cf[(size_t)mr * ldc + ncol] = v;
        if (Cbf) Cbf[(size_t)mr * ldc + ncol] = f2bf(v);
      }
    }
  }
}

// ---------------------------------------------------------------- fused LSTM step, LDS-staged weights
// One block per MEM tile (nt): 8 waves cover all 256 batch rows (MB=2 tiles each).
// Per 64-deep K slab, the 64 gate rows of W are staged in LDS once and shared by all waves.
__global__ void lstm_step_k(const u16* __restrict__ xbf,     // BS x EMB
                            const u16* __restrict__ hbf_in,  // BS x MEM
                            const u16* __restrict__ Wih,     // 4*MEM x EMB
                            const u16* __restrict__ Whh,     // 4*MEM x MEM
                            const float* __restrict__ b_ih,
                            const float* __restrict__ b_hh,
                            float* __restrict__ cstate,      // BS x MEM (in/out, fp32)
                            u16* __restrict__ hbf_out) {     // BS x MEM
  __shared__ __align__(16) u16 Bs[64 * LSTRIDE];             // 9 KB
  int nt = blockIdx.x;                 // 0..63
  int w = threadIdx.x >> 5;            // wave 0..7 -> M tiles 2w, 2w+1
  int lane = threadIdx.x & 31;
  int half = lane >> 4;
  int l = lane & 15;

  v8f zero = {};
  v8f acc[2][4];
#pragma unroll
  for (int i = 0; i < 2; ++i)
#pragma unroll
    for (int g = 0; g < 4; ++g) acc[i][g] = zero;

  // ---- phase 1: x @ Wih^T (K = EMB)
  for (int k0 = 0; k0 < EMB; k0 += 64) {
    for (int i = threadIdx.x; i < 64 * 8; i += 256) {        // 64 rows x 8 uint4
      int row = i >> 3, c8 = i & 7;
      int g = row >> 4;
      int wrow = g * MEM + nt * 16 + (row & 15);
      const uint4* gp = (const uint4*)(Wih + (size_t)wrow * EMB + k0 + c8 * 8);
      *(uint4*)&Bs[row * LSTRIDE + c8 * 8] = *gp;
    }
    __syncthreads();
#pragma unroll
    for (int kk = 0; kk < 64; kk += 32) {
      Frag a0, a1;
      const uint4* pa0 = (const uint4*)(xbf + (size_t)((2 * w) * 16 + l) * EMB + k0 + kk + half * 8);
      a0.u[0] = pa0[0]; a0.u[1] = pa0[2];
      const uint4* pa1 = (const uint4*)(xbf + (size_t)((2 * w + 1) * 16 + l) * EMB + k0 + kk + half * 8);
      a1.u[0] = pa1[0]; a1.u[1] = pa1[2];
#pragma unroll
      for (int g = 0; g < 4; ++g) {
        Frag b;
        const uint4* pb = (const uint4*)&Bs[(g * 16 + l) * LSTRIDE + kk + half * 8];
        b.u[0] = pb[0]; b.u[1] = pb[2];
        acc[0][g] = wmma_bf16(a0, b, acc[0][g]);
        acc[1][g] = wmma_bf16(a1, b, acc[1][g]);
      }
    }
    __syncthreads();
  }

  // ---- phase 2: h @ Whh^T (K = MEM)
  for (int k0 = 0; k0 < MEM; k0 += 64) {
    for (int i = threadIdx.x; i < 64 * 8; i += 256) {
      int row = i >> 3, c8 = i & 7;
      int g = row >> 4;
      int wrow = g * MEM + nt * 16 + (row & 15);
      const uint4* gp = (const uint4*)(Whh + (size_t)wrow * MEM + k0 + c8 * 8);
      *(uint4*)&Bs[row * LSTRIDE + c8 * 8] = *gp;
    }
    __syncthreads();
#pragma unroll
    for (int kk = 0; kk < 64; kk += 32) {
      Frag a0, a1;
      const uint4* pa0 = (const uint4*)(hbf_in + (size_t)((2 * w) * 16 + l) * MEM + k0 + kk + half * 8);
      a0.u[0] = pa0[0]; a0.u[1] = pa0[2];
      const uint4* pa1 = (const uint4*)(hbf_in + (size_t)((2 * w + 1) * 16 + l) * MEM + k0 + kk + half * 8);
      a1.u[0] = pa1[0]; a1.u[1] = pa1[2];
#pragma unroll
      for (int g = 0; g < 4; ++g) {
        Frag b;
        const uint4* pb = (const uint4*)&Bs[(g * 16 + l) * LSTRIDE + kk + half * 8];
        b.u[0] = pb[0]; b.u[1] = pb[2];
        acc[0][g] = wmma_bf16(a0, b, acc[0][g]);
        acc[1][g] = wmma_bf16(a1, b, acc[1][g]);
      }
    }
    __syncthreads();
  }

  // ---- pointwise LSTM cell
  int ncol = nt * 16 + l;
  float bi  = b_ih[ncol]           + b_hh[ncol];
  float bf_ = b_ih[MEM + ncol]     + b_hh[MEM + ncol];
  float bg  = b_ih[2 * MEM + ncol] + b_hh[2 * MEM + ncol];
  float bo  = b_ih[3 * MEM + ncol] + b_hh[3 * MEM + ncol];
#pragma unroll
  for (int i = 0; i < 2; ++i) {
#pragma unroll
    for (int j = 0; j < 8; ++j) {
      int mr = (2 * w + i) * 16 + j + 8 * half;
      size_t idx = (size_t)mr * MEM + ncol;
      float ig = sigmoidf_(acc[i][0][j] + bi);
      float fg = sigmoidf_(acc[i][1][j] + bf_);
      float gg = tanhf(acc[i][2][j] + bg);
      float og = sigmoidf_(acc[i][3][j] + bo);
      float cn = fg * cstate[idx] + ig * gg;
      float hn = og * tanhf(cn);
      cstate[idx] = cn;
      hbf_out[idx] = f2bf(hn);
    }
  }
}

// ---------------------------------------------------------------- logits GEMM, LDS-staged W_out
// One block per 80-column group (5 N tiles): 8 waves cover all 256 batch rows (MB=2).
__global__ void logits_k(const u16* __restrict__ hbf,       // BS x MEM
                         const u16* __restrict__ Wout,      // VOCAB x MEM
                         const float* __restrict__ bias,
                         float* __restrict__ Cf) {          // BS x VOCAB
  __shared__ __align__(16) u16 Bs[80 * LSTRIDE];            // 11.25 KB
  int ng = blockIdx.x;                 // 0..124 -> cols 80*ng..80*ng+79
  int w = threadIdx.x >> 5;
  int lane = threadIdx.x & 31;
  int half = lane >> 4;
  int l = lane & 15;

  v8f zero = {};
  v8f acc[2][NTILE_OUT];
#pragma unroll
  for (int i = 0; i < 2; ++i)
#pragma unroll
    for (int j = 0; j < NTILE_OUT; ++j) acc[i][j] = zero;

  for (int k0 = 0; k0 < MEM; k0 += 64) {
    for (int i = threadIdx.x; i < 80 * 8; i += 256) {        // 80 rows x 8 uint4
      int row = i >> 3, c8 = i & 7;
      const uint4* gp = (const uint4*)(Wout + (size_t)(ng * 80 + row) * MEM + k0 + c8 * 8);
      *(uint4*)&Bs[row * LSTRIDE + c8 * 8] = *gp;
    }
    __syncthreads();
#pragma unroll
    for (int kk = 0; kk < 64; kk += 32) {
      Frag a0, a1;
      const uint4* pa0 = (const uint4*)(hbf + (size_t)((2 * w) * 16 + l) * MEM + k0 + kk + half * 8);
      a0.u[0] = pa0[0]; a0.u[1] = pa0[2];
      const uint4* pa1 = (const uint4*)(hbf + (size_t)((2 * w + 1) * 16 + l) * MEM + k0 + kk + half * 8);
      a1.u[0] = pa1[0]; a1.u[1] = pa1[2];
#pragma unroll
      for (int j = 0; j < NTILE_OUT; ++j) {
        Frag b;
        const uint4* pb = (const uint4*)&Bs[(j * 16 + l) * LSTRIDE + kk + half * 8];
        b.u[0] = pb[0]; b.u[1] = pb[2];
        acc[0][j] = wmma_bf16(a0, b, acc[0][j]);
        acc[1][j] = wmma_bf16(a1, b, acc[1][j]);
      }
    }
    __syncthreads();
  }

#pragma unroll
  for (int j = 0; j < NTILE_OUT; ++j) {
    int ncol = ng * 80 + j * 16 + l;
    float bb = bias[ncol];
#pragma unroll
    for (int i = 0; i < 2; ++i) {
#pragma unroll
      for (int r = 0; r < 8; ++r) {
        int mr = (2 * w + i) * 16 + r + 8 * half;
        Cf[(size_t)mr * VOCAB + ncol] = acc[i][j][r] + bb;
      }
    }
  }
}

// ---------------------------------------------------------------- per-row log-softmax NLL
__global__ void loss_k(const float* __restrict__ logits, const int* __restrict__ cap,
                       int t, float* __restrict__ accum) {
  int b = blockIdx.x;
  const float* row = logits + (size_t)b * VOCAB;
  __shared__ float red[256];
  float m = -INFINITY;
  for (int i = threadIdx.x; i < VOCAB; i += 256) m = fmaxf(m, row[i]);
  red[threadIdx.x] = m;
  __syncthreads();
  for (int s = 128; s > 0; s >>= 1) {
    if (threadIdx.x < s) red[threadIdx.x] = fmaxf(red[threadIdx.x], red[threadIdx.x + s]);
    __syncthreads();
  }
  m = red[0];
  __syncthreads();
  float sum = 0.0f;
  for (int i = threadIdx.x; i < VOCAB; i += 256) sum += __expf(row[i] - m);
  red[threadIdx.x] = sum;
  __syncthreads();
  for (int s = 128; s > 0; s >>= 1) {
    if (threadIdx.x < s) red[threadIdx.x] += red[threadIdx.x + s];
    __syncthreads();
  }
  if (threadIdx.x == 0) {
    int tgt = cap[b * TLEN + t + 1];
    if (tgt != 0) {                     // PAD_IDX mask
      float lse = m + __logf(red[0]);
      float nll = lse - row[tgt];
      atomicAdd(&accum[0], nll);
      atomicAdd(&accum[1], 1.0f);
    }
  }
}

__global__ void finalize_k(const float* __restrict__ accum, float* __restrict__ out) {
  if (threadIdx.x == 0 && blockIdx.x == 0)
    out[0] = accum[0] / fmaxf(accum[1], 1.0f);
}

// ---------------------------------------------------------------- host
extern "C" void kernel_launch(void* const* d_in, const int* in_sizes, int n_in,
                              void* d_out, int out_size, void* d_ws, size_t ws_size,
                              hipStream_t stream) {
  const float* feature   = (const float*)d_in[0];
  const int*   caption   = (const int*)d_in[1];
  // d_in[2] = length (unused by the reference computation)
  const float* W_init_h  = (const float*)d_in[3];
  const float* b_init_h  = (const float*)d_in[4];
  const float* W_init_c  = (const float*)d_in[5];
  const float* b_init_c  = (const float*)d_in[6];
  const float* emb_table = (const float*)d_in[7];
  const float* W_ih      = (const float*)d_in[8];
  const float* b_ih      = (const float*)d_in[9];
  const float* W_hh      = (const float*)d_in[10];
  const float* b_hh      = (const float*)d_in[11];
  const float* W_out     = (const float*)d_in[12];
  const float* b_out     = (const float*)d_in[13];

  char* ws = (char*)d_ws;
  size_t off = 0;
  auto alloc = [&](size_t bytes) -> void* {
    void* p = ws + off;
    off += (bytes + 255) & ~(size_t)255;
    return p;
  };
  float* accum  = (float*)alloc(2 * sizeof(float));
  u16* wh_bf    = (u16*)alloc((size_t)MEM * FEAT * 2);
  u16* wc_bf    = (u16*)alloc((size_t)MEM * FEAT * 2);
  u16* wih_bf   = (u16*)alloc((size_t)4 * MEM * EMB * 2);
  u16* whh_bf   = (u16*)alloc((size_t)4 * MEM * MEM * 2);
  u16* wout_bf  = (u16*)alloc((size_t)VOCAB * MEM * 2);
  u16* feat_bf  = (u16*)alloc((size_t)BS * FEAT * 2);
  u16* x_bf     = (u16*)alloc((size_t)NSTEP * BS * EMB * 2);
  float* c_f32  = (float*)alloc((size_t)BS * MEM * 4);
  u16* h_bf0    = (u16*)alloc((size_t)BS * MEM * 2);
  u16* h_bf1    = (u16*)alloc((size_t)BS * MEM * 2);
  float* logits = (float*)alloc((size_t)BS * VOCAB * 4);

  zero_k<<<1, 32, 0, stream>>>(accum, 2);

  f32_to_bf16_k<<<512,  256, 0, stream>>>(feature,  feat_bf, BS * FEAT);
  f32_to_bf16_k<<<1024, 256, 0, stream>>>(W_init_h, wh_bf,   MEM * FEAT);
  f32_to_bf16_k<<<1024, 256, 0, stream>>>(W_init_c, wc_bf,   MEM * FEAT);
  f32_to_bf16_k<<<1024, 256, 0, stream>>>(W_ih,     wih_bf,  4 * MEM * EMB);
  f32_to_bf16_k<<<2048, 256, 0, stream>>>(W_hh,     whh_bf,  4 * MEM * MEM);
  f32_to_bf16_k<<<4096, 256, 0, stream>>>(W_out,    wout_bf, VOCAB * MEM);
  gather_x_k<<<2048, 256, 0, stream>>>(emb_table, caption, x_bf, NSTEP * BS * EMB);

  // h0 (bf16 for recurrence), c0 (fp32 cell state); run once, register-blocked path.
  gemm_bf16_k<2, 4><<<16, 256, 0, stream>>>(feat_bf, FEAT, wh_bf, FEAT, b_init_h,
                                            nullptr, h_bf0, MEM, 8, 16, FEAT);
  gemm_bf16_k<2, 4><<<16, 256, 0, stream>>>(feat_bf, FEAT, wc_bf, FEAT, b_init_c,
                                            c_f32, nullptr, MEM, 8, 16, FEAT);

  u16* hcur = h_bf0;
  u16* hnext = h_bf1;
  for (int t = 0; t < NSTEP; ++t) {
    lstm_step_k<<<64, 256, 0, stream>>>(x_bf + (size_t)t * BS * EMB, hcur,
                                        wih_bf, whh_bf, b_ih, b_hh, c_f32, hnext);
    logits_k<<<125, 256, 0, stream>>>(hnext, wout_bf, b_out, logits);
    loss_k<<<BS, 256, 0, stream>>>(logits, caption, t, accum);
    u16* tmp = hcur; hcur = hnext; hnext = tmp;
  }

  finalize_k<<<1, 1, 0, stream>>>(accum, (float*)d_out);
}
